// AUGuidedAttentionFusion_64639257805106
// MI455X (gfx1250) — matrix operand order repaired and checked by reading, not compiled, not run on hardware
//
#include <hip/hip_runtime.h>
#include <hip/hip_bf16.h>

// ---------------------------------------------------------------------------
// AU-guided attention fusion, MI455X (gfx1250, wave32, WMMA bf16 16x16x32).
// B=32768 rows, D=768, A=17, H=128, NH=4, DH=32.
//
// Heavy GEMMs (q-proj, attn_out, attn_feat, gate) run on v_wmma_f32_16x16x32_bf16.
// k/v projections are algebraically folded: au_tokens = LN(s*w_tok + c_a) is a
// function of scalar s=au_feat[b,a], so k[b,a,:] = (s*U1k + U2k[a])*isig + U3k.
// q-proj folds two matmuls: q = vf @ (wq @ w_q).T + bqq.
// ---------------------------------------------------------------------------

typedef __bf16 bf16;
typedef bf16  v16bf __attribute__((ext_vector_type(16)));
typedef float v8f   __attribute__((ext_vector_type(8)));

#define Bn 32768
#define Dn 768
#define An 17
#define Hn 128

// ---- workspace layout (bytes) --------------------------------------------
#define OFF_WQQ    ((size_t)0)                    // bf16 H x D      196608
#define OFF_WAO    ((size_t)196608)               // bf16 H x H       32768
#define OFF_WOUT   ((size_t)229376)               // bf16 D x H      196608
#define OFF_WGATE  ((size_t)425984)               // bf16 D x 2D   2359296
#define OFF_U1K    ((size_t)2785280)              // f32 H
#define OFF_U3K    ((size_t)2785792)
#define OFF_U1V    ((size_t)2786304)
#define OFF_U3V    ((size_t)2786816)
#define OFF_U2K    ((size_t)2787328)              // f32 A*H         8704
#define OFF_U2V    ((size_t)2796032)
#define OFF_BQQ    ((size_t)2804736)              // f32 H
#define OFF_COV    ((size_t)2805248)              // f32 A (pad)
#define OFF_VC     ((size_t)2805376)
#define OFF_CONSTS ((size_t)2805504)              // f32: [0]=Vw
// total ~2.81 MB of d_ws

// ---- WMMA fragment helpers (CDNA5 layouts, wave32) ------------------------
// A 16x32 bf16: lanes 0-15 row M=lane hold K {g*16..g*16+15}, g=lane>>4
__device__ __forceinline__ v16bf lda_frag(const bf16* __restrict__ tile, int ld,
                                          int kBase, int lane) {
  const int row = lane & 15, g = (lane >> 4) & 1;
  const bf16* p = tile + row * ld + kBase + g * 16;
  v16bf a;
#pragma unroll
  for (int e = 0; e < 8; ++e) a[e] = p[e];
#pragma unroll
  for (int e = 0; e < 8; ++e) a[e + 8] = p[8 + e];
  return a;
}

// B 32x16 bf16 from row-major weight W[N][K] (computing X @ W^T):
// lane n=lane&15 column, lanes 0-15 K=0..15, lanes 16-31 K=16..31.
__device__ __forceinline__ v16bf ldb_frag(const bf16* __restrict__ W, int ldw,
                                          int nBase, int kBase, int lane) {
  const int n = lane & 15, g = (lane >> 4) & 1;
  const bf16* p = W + (size_t)(nBase + n) * ldw + kBase + g * 16;
  v16bf b;
#pragma unroll
  for (int e = 0; e < 16; ++e) b[e] = p[e];
  return b;
}

__device__ __forceinline__ v8f wmma_bf16(v16bf a, v16bf b, v8f c) {
  return __builtin_amdgcn_wmma_f32_16x16x32_bf16(false, a, false, b,
                                                 (short)0, c, false, false);
}

__device__ __forceinline__ float wred_sum(float x) {
#pragma unroll
  for (int o = 16; o >= 1; o >>= 1) x += __shfl_xor(x, o, 32);
  return x;
}
__device__ __forceinline__ float wred_max(float x) {
#pragma unroll
  for (int o = 16; o >= 1; o >>= 1) x = fmaxf(x, __shfl_xor(x, o, 32));
  return x;
}

// ---------------------------------------------------------------------------
// prep kernels (run once per launch; trivial cost vs 129 GFLOP main pass)
// ---------------------------------------------------------------------------
__global__ void prep_small(const float* __restrict__ w_tok, const float* __restrict__ b_tok,
                           const float* __restrict__ pos, const float* __restrict__ g_auln,
                           const float* __restrict__ b_auln, const float* __restrict__ w_in,
                           const float* __restrict__ b_in, const float* __restrict__ b_q,
                           float* U1k, float* U3k, float* U1v, float* U3v,
                           float* U2k, float* U2v, float* bqq,
                           float* Cov, float* Vc, float* consts) {
  __shared__ float wtil[Hn];
  __shared__ float gg[Hn];
  __shared__ float ctil[An][Hn];
  int t = threadIdx.x;
  if (t < Hn) { wtil[t] = w_tok[t]; gg[t] = g_auln[t]; }
  __syncthreads();
  if (t == 0) {
    float mw = 0.f;
    for (int j = 0; j < Hn; ++j) mw += wtil[j];
    mw *= (1.f / Hn);
    float vw = 0.f;
    for (int j = 0; j < Hn; ++j) { wtil[j] -= mw; vw += wtil[j] * wtil[j]; }
    consts[0] = vw * (1.f / Hn);
  }
  __syncthreads();
  if (t < An) {
    float mc = 0.f;
    for (int j = 0; j < Hn; ++j) { float c = b_tok[j] + pos[t * Hn + j]; ctil[t][j] = c; mc += c; }
    mc *= (1.f / Hn);
    float vc = 0.f, cv = 0.f;
    for (int j = 0; j < Hn; ++j) {
      float d = ctil[t][j] - mc; ctil[t][j] = d; vc += d * d; cv += d * wtil[j];
    }
    Vc[t] = vc * (1.f / Hn); Cov[t] = cv * (1.f / Hn);
  }
  __syncthreads();
  if (t < Hn) {
    const float* wk = w_in + Hn * Hn;
    const float* wv = w_in + 2 * Hn * Hn;
    float u1k = 0, u3k = 0, u1v = 0, u3v = 0, bq2 = 0;
    for (int j = 0; j < Hn; ++j) {
      float k = wk[t * Hn + j], v = wv[t * Hn + j];
      float wg = gg[j] * wtil[j];
      u1k += k * wg;          u1v += v * wg;
      u3k += k * b_auln[j];   u3v += v * b_auln[j];
      bq2 += w_in[t * Hn + j] * b_q[j];
    }
    U1k[t] = u1k;             U1v[t] = u1v;
    U3k[t] = u3k + b_in[Hn + t];
    U3v[t] = u3v + b_in[2 * Hn + t];
    bqq[t] = bq2 + b_in[t];
  }
  __syncthreads();
  for (int idx = t; idx < An * Hn; idx += blockDim.x) {
    int a = idx >> 7, h = idx & (Hn - 1);
    const float* wk = w_in + Hn * Hn + h * Hn;
    const float* wv = w_in + 2 * Hn * Hn + h * Hn;
    float u2k = 0, u2v = 0;
    for (int j = 0; j < Hn; ++j) { float cg = gg[j] * ctil[a][j]; u2k += wk[j] * cg; u2v += wv[j] * cg; }
    U2k[idx] = u2k; U2v[idx] = u2v;
  }
}

__global__ void prep_wqq(const float* __restrict__ w_in, const float* __restrict__ w_q,
                         bf16* __restrict__ Wqq) {
  int idx = blockIdx.x * blockDim.x + threadIdx.x;
  if (idx >= Hn * Dn) return;
  int h = idx / Dn, d = idx % Dn;
  float acc = 0.f;
  for (int j = 0; j < Hn; ++j) acc += w_in[h * Hn + j] * w_q[(size_t)j * Dn + d];
  Wqq[idx] = (bf16)acc;
}

__global__ void prep_cast(const float* __restrict__ w_ao, const float* __restrict__ w_out,
                          const float* __restrict__ w_gate, bf16* __restrict__ wao_bf,
                          bf16* __restrict__ wout_bf, bf16* __restrict__ wgate_bf) {
  int i = blockIdx.x * blockDim.x + threadIdx.x;
  int stride = gridDim.x * blockDim.x;
  for (int k = i; k < Hn * Hn; k += stride) wao_bf[k] = (bf16)w_ao[k];
  for (int k = i; k < Dn * Hn; k += stride) wout_bf[k] = (bf16)w_out[k];
  for (int k = i; k < Dn * 2 * Dn; k += stride) wgate_bf[k] = (bf16)w_gate[k];
}

// ---------------------------------------------------------------------------
// main fused kernel: 16 rows per block, 8 waves of 32
// ---------------------------------------------------------------------------
__global__ __launch_bounds__(256, 2)
void fused_main(const float* __restrict__ visual, const float* __restrict__ au,
                const bf16* __restrict__ Wqq, const float* __restrict__ bqq,
                const float* __restrict__ U1k, const float* __restrict__ U3k,
                const float* __restrict__ U1v, const float* __restrict__ U3v,
                const float* __restrict__ U2k, const float* __restrict__ U2v,
                const float* __restrict__ Cov, const float* __restrict__ Vc,
                const float* __restrict__ consts,
                const bf16* __restrict__ wao_bf, const float* __restrict__ b_ao,
                const float* __restrict__ g_aln, const float* __restrict__ b_aln,
                const bf16* __restrict__ wout_bf, const float* __restrict__ b_out,
                const bf16* __restrict__ wgate_bf, const float* __restrict__ b_gate,
                float* __restrict__ out) {
  __shared__ bf16 vf_bf[16 * Dn];                  // 24576 B, live S1..S8
  __shared__ bf16 aln_bf[16 * Hn];                 //  4096 B, live S6..S7
  __shared__ __align__(16) unsigned char smemX[16 * Dn * 2];  // 24576 B union
  float* rowbuf = (float*)smemX;                   // [16][128] f32  (q, then attn_out)
  bf16*  ctx_bf = (bf16*)(smemX + 8192);           // [16][128] bf16
  float* attn_l = (float*)(smemX + 12288);         // [16][4][20] f32
  float* isig_l = (float*)(smemX + 17408);         // [16][20]   f32
  bf16*  af_bf  = (bf16*)smemX;                    // [16][768] bf16, overlays after S6

  const int tid  = threadIdx.x;
  const int wave = tid >> 5;
  const int lane = tid & 31;
  const int row0 = blockIdx.x * 16;
  const size_t vbase = (size_t)row0 * Dn;

  // ---- S1: stage visual_feat tile as bf16 --------------------------------
  for (int i = tid; i < 16 * Dn; i += 256) vf_bf[i] = (bf16)visual[vbase + i];

  // ---- S2: per (row, a) inverse sigma of the folded token layernorm ------
  {
    float Vw = consts[0];
    for (int i = tid; i < 16 * An; i += 256) {
      int r = i / An, a = i % An;
      float s = au[(size_t)(row0 + r) * An + a];
      float sig2 = s * s * Vw + 2.f * s * Cov[a] + Vc[a] + 1e-5f;
      isig_l[r * 20 + a] = rsqrtf(sig2);
    }
  }
  __syncthreads();

  // ---- S3: q = vf @ Wqq^T + bqq   (M16, N128, K768) -----------------------
  {
    v8f acc = {};
    for (int kt = 0; kt < Dn / 32; ++kt) {
      v16bf a = lda_frag(vf_bf, Dn, kt * 32, lane);
      v16bf b = ldb_frag(Wqq, Dn, wave * 16, kt * 32, lane);
      acc = wmma_bf16(a, b, acc);
    }
    int n = wave * 16 + (lane & 15);
    float bias = bqq[n];
#pragma unroll
    for (int e = 0; e < 8; ++e) {
      int m = ((lane >> 4) << 3) + e;
      rowbuf[m * Hn + n] = acc[e] + bias;
    }
  }
  __syncthreads();

  // ---- S4: attention (2 rows per wave, head == one 32-lane chunk) --------
  {
    const float scale = 0.17677669529663688f;  // 1/sqrt(32)
#pragma unroll
    for (int rr = 0; rr < 2; ++rr) {
      int row = wave + rr * 8;
      size_t arow = (size_t)(row0 + row) * An;
      // scores: lane = a
      int a = lane;
      bool act = a < An;
      int aa = act ? a : 0;
      float s = au[arow + aa];
      float isg = isig_l[row * 20 + aa];
      float sc[4] = {0.f, 0.f, 0.f, 0.f};
#pragma unroll
      for (int c = 0; c < 4; ++c) {
        float acc = 0.f;
        for (int h = c * 32; h < c * 32 + 32; ++h) {
          float kv = (s * U1k[h] + U2k[aa * Hn + h]) * isg + U3k[h];
          acc += rowbuf[row * Hn + h] * kv;
        }
        sc[c] = acc * scale;
      }
#pragma unroll
      for (int c = 0; c < 4; ++c) {
        float x = act ? sc[c] : -1e30f;
        float mx = wred_max(x);
        float e = act ? __expf(x - mx) : 0.f;
        float sm = wred_sum(e);
        if (act) attn_l[(row * 4 + c) * 20 + a] = e / sm;
      }
      __builtin_amdgcn_s_barrier_signal(-1);  // lightweight sync not needed per-wave; wave-local
      __builtin_amdgcn_s_barrier_wait(-1);
      // ctx: lane = h within chunk c
#pragma unroll
      for (int c = 0; c < 4; ++c) {
        int h = c * 32 + lane;
        float u1 = U1v[h], u3 = U3v[h];
        float acc = 0.f;
        for (int a2 = 0; a2 < An; ++a2) {
          float s2 = au[arow + a2];
          float ig = isig_l[row * 20 + a2];
          float vv = (s2 * u1 + U2v[a2 * Hn + h]) * ig + u3;
          acc += attn_l[(row * 4 + c) * 20 + a2] * vv;
        }
        ctx_bf[row * Hn + h] = (bf16)acc;
      }
    }
  }
  __syncthreads();

  // ---- S5: attn_out = ctx @ w_ao^T + b_ao  (M16, N128, K128) --------------
  {
    v8f acc = {};
    for (int kt = 0; kt < Hn / 32; ++kt) {
      v16bf a = lda_frag(ctx_bf, Hn, kt * 32, lane);
      v16bf b = ldb_frag(wao_bf, Hn, wave * 16, kt * 32, lane);
      acc = wmma_bf16(a, b, acc);
    }
    int n = wave * 16 + (lane & 15);
    float bias = b_ao[n];
#pragma unroll
    for (int e = 0; e < 8; ++e) {
      int m = ((lane >> 4) << 3) + e;
      rowbuf[m * Hn + n] = acc[e] + bias;
    }
  }
  __syncthreads();

  // ---- S6: layernorm(attn_out) -> aln_bf ----------------------------------
  {
#pragma unroll
    for (int rr = 0; rr < 2; ++rr) {
      int row = wave + rr * 8;
      float x[4], sum = 0.f;
#pragma unroll
      for (int c = 0; c < 4; ++c) { x[c] = rowbuf[row * Hn + c * 32 + lane]; sum += x[c]; }
      float mu = wred_sum(sum) * (1.f / Hn);
      float vs = 0.f;
#pragma unroll
      for (int c = 0; c < 4; ++c) { float d = x[c] - mu; vs += d * d; }
      float inv = rsqrtf(wred_sum(vs) * (1.f / Hn) + 1e-5f);
#pragma unroll
      for (int c = 0; c < 4; ++c) {
        int h = c * 32 + lane;
        aln_bf[row * Hn + h] = (bf16)((x[c] - mu) * inv * g_aln[h] + b_aln[h]);
      }
    }
  }
  __syncthreads();

  // ---- S7: attn_feat = aln @ w_out^T + b_out  (M16, N768, K128) -----------
  // af_bf overlays rowbuf/ctx/attn/isig (all dead now).
  {
    for (int t = 0; t < 6; ++t) {
      int nBase = (wave + t * 8) * 16;
      v8f acc = {};
      for (int kt = 0; kt < Hn / 32; ++kt) {
        v16bf a = lda_frag(aln_bf, Hn, kt * 32, lane);
        v16bf b = ldb_frag(wout_bf, Hn, nBase, kt * 32, lane);
        acc = wmma_bf16(a, b, acc);
      }
      int n = nBase + (lane & 15);
      float bias = b_out[n];
#pragma unroll
      for (int e = 0; e < 8; ++e) {
        int m = ((lane >> 4) << 3) + e;
        af_bf[m * Dn + n] = (bf16)(acc[e] + bias);
      }
    }
  }
  __syncthreads();

  // ---- S8: gate GEMM (M16, N768, K1536) + sigmoid + blend -----------------
  {
    for (int t = 0; t < 6; ++t) {
      int nBase = (wave + t * 8) * 16;
      v8f acc = {};
      for (int kt = 0; kt < 48; ++kt) {
        v16bf a = (kt < 24) ? lda_frag(vf_bf, Dn, kt * 32, lane)
                            : lda_frag(af_bf, Dn, (kt - 24) * 32, lane);
        v16bf b = ldb_frag(wgate_bf, 2 * Dn, nBase, kt * 32, lane);
        acc = wmma_bf16(a, b, acc);
      }
      int n = nBase + (lane & 15);
      float bg = b_gate[n];
#pragma unroll
      for (int e = 0; e < 8; ++e) {
        int m = ((lane >> 4) << 3) + e;
        float g = 1.f / (1.f + __expf(-(acc[e] + bg)));
        float af = (float)af_bf[m * Dn + n];
        float vf = visual[vbase + (size_t)m * Dn + n];
        out[vbase + (size_t)m * Dn + n] = g * af + (1.f - g) * vf;
      }
    }
  }
}

// ---------------------------------------------------------------------------
extern "C" void kernel_launch(void* const* d_in, const int* in_sizes, int n_in,
                              void* d_out, int out_size, void* d_ws, size_t ws_size,
                              hipStream_t stream) {
  const float* visual = (const float*)d_in[0];
  const float* au     = (const float*)d_in[1];
  const float* w_tok  = (const float*)d_in[2];
  const float* b_tok  = (const float*)d_in[3];
  const float* pos    = (const float*)d_in[4];
  const float* g_auln = (const float*)d_in[5];
  const float* b_auln = (const float*)d_in[6];
  const float* w_q    = (const float*)d_in[7];
  const float* b_q    = (const float*)d_in[8];
  const float* w_in   = (const float*)d_in[9];
  const float* b_in   = (const float*)d_in[10];
  const float* w_ao   = (const float*)d_in[11];
  const float* b_ao   = (const float*)d_in[12];
  const float* g_aln  = (const float*)d_in[13];
  const float* b_aln  = (const float*)d_in[14];
  const float* w_out  = (const float*)d_in[15];
  const float* b_out  = (const float*)d_in[16];
  const float* w_gate = (const float*)d_in[17];
  const float* b_gate = (const float*)d_in[18];
  float* outp = (float*)d_out;

  char* ws = (char*)d_ws;
  bf16*  Wqq    = (bf16*)(ws + OFF_WQQ);
  bf16*  waoB   = (bf16*)(ws + OFF_WAO);
  bf16*  woutB  = (bf16*)(ws + OFF_WOUT);
  bf16*  wgateB = (bf16*)(ws + OFF_WGATE);
  float* U1k    = (float*)(ws + OFF_U1K);
  float* U3k    = (float*)(ws + OFF_U3K);
  float* U1v    = (float*)(ws + OFF_U1V);
  float* U3v    = (float*)(ws + OFF_U3V);
  float* U2k    = (float*)(ws + OFF_U2K);
  float* U2v    = (float*)(ws + OFF_U2V);
  float* bqq    = (float*)(ws + OFF_BQQ);
  float* Cov    = (float*)(ws + OFF_COV);
  float* Vc     = (float*)(ws + OFF_VC);
  float* consts = (float*)(ws + OFF_CONSTS);

  prep_small<<<1, 256, 0, stream>>>(w_tok, b_tok, pos, g_auln, b_auln, w_in, b_in, b_q,
                                    U1k, U3k, U1v, U3v, U2k, U2v, bqq, Cov, Vc, consts);
  prep_wqq<<<(Hn * Dn + 255) / 256, 256, 0, stream>>>(w_in, w_q, Wqq);
  prep_cast<<<512, 256, 0, stream>>>(w_ao, w_out, w_gate, waoB, woutB, wgateB);
  fused_main<<<Bn / 16, 256, 0, stream>>>(visual, au, Wqq, bqq, U1k, U3k, U1v, U3v,
                                          U2k, U2v, Cov, Vc, consts, waoB, b_ao,
                                          g_aln, b_aln, woutB, b_out, wgateB, b_gate,
                                          outp);
}
